// WindowAttention_28312424415814
// MI455X (gfx1250) — matrix operand order repaired and checked by reading, not compiled, not run on hardware
//
#include <hip/hip_runtime.h>

typedef _Float16 h16;
typedef __attribute__((ext_vector_type(16))) _Float16 v16h;
typedef __attribute__((ext_vector_type(8)))  _Float16 v8h;
typedef __attribute__((ext_vector_type(8)))  float    v8f;
typedef __attribute__((ext_vector_type(4)))  float    v4f;

#define WMMA16(a, b, c) \
  __builtin_amdgcn_wmma_f32_16x16x32_f16(false, (a), false, (b), (short)0, (c), false, false)

// ---------- CDNA5 async global->LDS helpers ----------

// Wave-relative LDS byte offset of a shared-memory pointer (AS3 ptrtoint).
__device__ inline unsigned lds_offset(const void* p) {
  return (unsigned)(unsigned long long)(__attribute__((address_space(3))) const void*)p;
}

__device__ inline void wait_async0() {
  asm volatile("s_wait_asynccnt 0x0" ::: "memory");
}

// Cooperative weight-slice stage via GLOBAL_LOAD_ASYNC_TO_LDS_B128 (ASYNCcnt):
// 256 threads move W[k..k+31][0..255] (16 KB) as [n][k_local] so B-fragments
// are two 16B runs per lane. Thread t = output row n; 64 contiguous bytes per thread.
__device__ inline void stage_slice_async(const h16* __restrict__ src_base,  // W + kt*32
                                         h16* buf, int tid) {
  unsigned lds = lds_offset(buf + tid * 32);
  const h16* g = src_base + tid * 256;
  asm volatile(
      "global_load_async_to_lds_b128 %0, %1, off\n\t"
      "global_load_async_to_lds_b128 %0, %1, off offset:16\n\t"
      "global_load_async_to_lds_b128 %0, %1, off offset:32\n\t"
      "global_load_async_to_lds_b128 %0, %1, off offset:48"
      :: "v"(lds), "v"(g) : "memory");
}

// ---------- fragment loaders ----------

__device__ inline v16h cat16(v8h a, v8h b) {
  return __builtin_shufflevector(a, b, 0, 1, 2, 3, 4, 5, 6, 7,
                                       8, 9, 10, 11, 12, 13, 14, 15);
}

// A-operand (16x32 f16) from global f32 activations, row-major, per ISA layout:
// lane<16: K = kb+{0..7}, kb+{16..23}; lane>=16: +8. p already includes hi*8.
__device__ inline v16h frag_a_g(const float* p) {
  v4f a0 = *(const v4f*)(p);
  v4f a1 = *(const v4f*)(p + 4);
  v4f a2 = *(const v4f*)(p + 16);
  v4f a3 = *(const v4f*)(p + 20);
  v16h r;
#pragma unroll
  for (int i = 0; i < 4; ++i) {
    r[i] = (h16)a0[i]; r[4 + i] = (h16)a1[i];
    r[8 + i] = (h16)a2[i]; r[12 + i] = (h16)a3[i];
  }
  return r;
}

// A-operand from f16 LDS (16B-aligned): K{0..7} at p, K{16..23} at p+16.
__device__ inline v16h frag_a_lds(const _Float16* p) {
  return cat16(*(const v8h*)(p), *(const v8h*)(p + 16));
}

// B-operand from f16 LDS (16B-aligned): K{0..7} at p, K{16..23} at p+16.
__device__ inline v16h frag_b_lds(const _Float16* p) {
  return cat16(*(const v8h*)(p), *(const v8h*)(p + 16));
}

// 64x256 @ 256x256 GEMM with double-buffered async LDS weight staging.
// A either from global f32 (Ag) or LDS f16 row-major stride 256 (Al).
template <bool A_LDS>
__device__ inline void gemm256(const float* Ag, const _Float16* Al,
                               const h16* W, h16* sW0, h16* sW1,
                               v8f (&acc)[8], int tid, int lo, int hi, int ch) {
  stage_slice_async(W, sW0, tid);
  for (int kt = 0; kt < 8; ++kt) {
    wait_async0();    // this wave's slice-kt copies have landed in LDS
    __syncthreads();  // slice kt visible to all; kt+1 buffer free to overwrite
    const h16* cur = (kt & 1) ? sW1 : sW0;
    h16* nxt = (kt & 1) ? sW0 : sW1;
    if (kt < 7) stage_slice_async(W + (kt + 1) * 32, nxt, tid);
    v16h a = A_LDS ? frag_a_lds(Al + kt * 32) : frag_a_g(Ag + kt * 32);
#pragma unroll
    for (int nt = 0; nt < 8; ++nt) {
      int n = (ch * 8 + nt) * 16 + lo;
      v16h bf = frag_b_lds(cur + n * 32 + hi * 8);
      acc[nt] = WMMA16(a, bf, acc[nt]);
    }
  }
}

// ---------- prep: convert shared weights f32 -> f16 once per call ----------
// q_w: 65536, kv_w: 131072, proj_w: 65536 -> 262144 halves (512 KB) in d_ws.
__global__ __launch_bounds__(256) void convert_weights(
    const float* __restrict__ qw, const float* __restrict__ kvw,
    const float* __restrict__ pw, h16* __restrict__ w16)
{
  int i = blockIdx.x * 256 + threadIdx.x;  // 0 .. 262143
  float v;
  if (i < 65536)       v = qw[i];
  else if (i < 196608) v = kvw[i - 65536];
  else                 v = pw[i - 196608];
  w16[i] = (h16)v;
}

// ---------- fused SwinV2 window attention: one workgroup per window ----------
__global__ __launch_bounds__(256) void swinv2_window_attn(
    const float* __restrict__ x,   const float* __restrict__ c,
    const float* __restrict__ mask,
    const h16* __restrict__ w16,                       // [qw | kvw | pw] as f16
    const float* __restrict__ qb,  const float* __restrict__ kvb,
    const float* __restrict__ lsc, const float* __restrict__ rpb,
    const float* __restrict__ pb,
    float* __restrict__ out)
{
  // 160 KB static LDS: 128 KB data pool (sP aliases dead sQ+sK) + 32 KB
  // double-buffered weight staging. 2 blocks per 320 KB WGP.
  __shared__ __align__(32) unsigned char pool[163840];
  _Float16* sQ  = (_Float16*)(pool);            // 64x256 f16 normalized q
  _Float16* sK  = (_Float16*)(pool + 32768);    // 64x256 f16 normalized k
  _Float16* sVT = (_Float16*)(pool + 65536);    // 256x64 f16 v transposed [d][token]
  _Float16* sO  = (_Float16*)(pool + 98304);    // 64x256 f16 attention output
  _Float16* sP  = (_Float16*)(pool);            // 8 x (64x64) f16 softmax probs
  h16* sW0 = (h16*)(pool + 131072);             // 16 KB weight slice buf 0
  h16* sW1 = (h16*)(pool + 147456);             // 16 KB weight slice buf 1

  const h16* wq  = w16;            // 256x256
  const h16* wkv = w16 + 65536;    // 512x256
  const h16* wp  = w16 + 196608;   // 256x256

  const int b    = blockIdx.x;
  const int tid  = threadIdx.x;
  const int wave = tid >> 5;
  const int lane = tid & 31;
  const int lo   = lane & 15;
  const int hi   = (lane >> 4) & 1;
  const int rs   = wave & 3;   // 16-row strip
  const int ch   = wave >> 2;  // column half (8 n-tiles of 16)

  const float* Ax = x + ((size_t)b * 64 + rs * 16 + lo) * 256 + hi * 8;
  const float* Ac = c + ((size_t)b * 64 + rs * 16 + lo) * 256 + hi * 8;

  // ---------------- Q projection + cosine normalize -> sQ ----------------
  {
    v8f acc[8];
#pragma unroll
    for (int t = 0; t < 8; ++t) acc[t] = v8f{};
    gemm256<false>(Ax, nullptr, wq, sW0, sW1, acc, tid, lo, hi, ch);
#pragma unroll
    for (int j = 0; j < 4; ++j) {            // 4 heads per wave, 2 tiles each
      int cb = ch * 128 + j * 32;
      float b0 = qb[cb + lo], b1 = qb[cb + 16 + lo];
#pragma unroll
      for (int r = 0; r < 8; ++r) {
        float a0 = acc[2 * j][r] + b0;
        float a1 = acc[2 * j + 1][r] + b1;
        float ss = a0 * a0 + a1 * a1;
        ss += __shfl_xor(ss, 1); ss += __shfl_xor(ss, 2);
        ss += __shfl_xor(ss, 4); ss += __shfl_xor(ss, 8);
        float inv = 1.0f / fmaxf(sqrtf(ss), 1e-12f);
        int M = rs * 16 + r + hi * 8;
        sQ[M * 256 + cb + lo]      = (h16)(a0 * inv);
        sQ[M * 256 + cb + 16 + lo] = (h16)(a1 * inv);
      }
    }
  }

  // ---------------- K projection (kv rows 0..255) + normalize -> sK ----------------
  {
    v8f acc[8];
#pragma unroll
    for (int t = 0; t < 8; ++t) acc[t] = v8f{};
    gemm256<false>(Ac, nullptr, wkv, sW0, sW1, acc, tid, lo, hi, ch);
#pragma unroll
    for (int j = 0; j < 4; ++j) {
      int cb = ch * 128 + j * 32;
      float b0 = kvb[cb + lo], b1 = kvb[cb + 16 + lo];
#pragma unroll
      for (int r = 0; r < 8; ++r) {
        float a0 = acc[2 * j][r] + b0;
        float a1 = acc[2 * j + 1][r] + b1;
        float ss = a0 * a0 + a1 * a1;
        ss += __shfl_xor(ss, 1); ss += __shfl_xor(ss, 2);
        ss += __shfl_xor(ss, 4); ss += __shfl_xor(ss, 8);
        float inv = 1.0f / fmaxf(sqrtf(ss), 1e-12f);
        int M = rs * 16 + r + hi * 8;
        sK[M * 256 + cb + lo]      = (h16)(a0 * inv);
        sK[M * 256 + cb + 16 + lo] = (h16)(a1 * inv);
      }
    }
  }

  // ---------------- V projection (kv rows 256..511) -> sVT (transposed) ----------------
  {
    v8f acc[8];
#pragma unroll
    for (int t = 0; t < 8; ++t) acc[t] = v8f{};
    gemm256<false>(Ac, nullptr, wkv + 65536, sW0, sW1, acc, tid, lo, hi, ch);
#pragma unroll
    for (int nt = 0; nt < 8; ++nt) {
      int col = (ch * 8 + nt) * 16 + lo;     // v channel 0..255
      float bv = kvb[256 + col];
      v8h pk;
#pragma unroll
      for (int r = 0; r < 8; ++r) pk[r] = (h16)(acc[nt][r] + bv);
      // tokens rs*16+hi*8 .. +7 are contiguous -> one 16B ds_store
      *(v8h*)(sVT + col * 64 + rs * 16 + hi * 8) = pk;
    }
  }

  __syncthreads();

  // ---------------- Attention: head h = wave ----------------
  {
    const int h = wave;
    v8f s[4][4];
#pragma unroll
    for (int mt = 0; mt < 4; ++mt) {
      v16h a = frag_a_lds(sQ + (mt * 16 + lo) * 256 + h * 32 + hi * 8);
#pragma unroll
      for (int nt = 0; nt < 4; ++nt) {
        v16h bf = frag_b_lds(sK + (nt * 16 + lo) * 256 + h * 32 + hi * 8);
        v8f z{};
        s[mt][nt] = WMMA16(a, bf, z);
      }
    }
    __syncthreads();  // sQ/sK reads complete; sP may now overwrite that region

    float ls = __expf(fminf(lsc[h], 4.605170186f));   // exp(min(logit_scale, ln 100))
    const float* mw = mask + (size_t)(b & 1023) * 4096;
    _Float16* sPh = sP + h * 4096;

#pragma unroll
    for (int mt = 0; mt < 4; ++mt) {
#pragma unroll
      for (int r = 0; r < 8; ++r) {
        int M = mt * 16 + r + hi * 8;
        int mi = M >> 3, mj = M & 7;
#pragma unroll
        for (int nt = 0; nt < 4; ++nt) {
          int Nn = nt * 16 + lo;
          int ni = Nn >> 3, nj = Nn & 7;
          int idx = (mi - ni + 7) * 15 + (mj - nj + 7);  // rel-pos index (15 = 2*Ww-1)
          s[mt][nt][r] = s[mt][nt][r] * ls + rpb[idx * 8 + h] + mw[M * 64 + Nn];
        }
        float mx = fmaxf(fmaxf(s[mt][0][r], s[mt][1][r]),
                         fmaxf(s[mt][2][r], s[mt][3][r]));
        mx = fmaxf(mx, __shfl_xor(mx, 1)); mx = fmaxf(mx, __shfl_xor(mx, 2));
        mx = fmaxf(mx, __shfl_xor(mx, 4)); mx = fmaxf(mx, __shfl_xor(mx, 8));
        float sum = 0.f;
#pragma unroll
        for (int nt = 0; nt < 4; ++nt) {
          float e = __expf(s[mt][nt][r] - mx);
          s[mt][nt][r] = e;
          sum += e;
        }
        sum += __shfl_xor(sum, 1); sum += __shfl_xor(sum, 2);
        sum += __shfl_xor(sum, 4); sum += __shfl_xor(sum, 8);
        float inv = 1.0f / sum;
#pragma unroll
        for (int nt = 0; nt < 4; ++nt)
          sPh[M * 64 + nt * 16 + lo] = (h16)(s[mt][nt][r] * inv);
      }
    }

    // ---- P @ V_h -> sO (in-wave LDS RAW; same-wave DS ops are in order) ----
    v8f o[4][2];
#pragma unroll
    for (int mt = 0; mt < 4; ++mt)
#pragma unroll
      for (int nt = 0; nt < 2; ++nt) o[mt][nt] = v8f{};
#pragma unroll
    for (int kt = 0; kt < 2; ++kt) {
#pragma unroll
      for (int mt = 0; mt < 4; ++mt) {
        v16h a = frag_a_lds(sPh + (mt * 16 + lo) * 64 + kt * 32 + hi * 8);
#pragma unroll
        for (int nt = 0; nt < 2; ++nt) {
          v16h bf = frag_b_lds(sVT + (h * 32 + nt * 16 + lo) * 64 + kt * 32 + hi * 8);
          o[mt][nt] = WMMA16(a, bf, o[mt][nt]);
        }
      }
    }
#pragma unroll
    for (int mt = 0; mt < 4; ++mt)
#pragma unroll
      for (int nt = 0; nt < 2; ++nt)
#pragma unroll
        for (int r = 0; r < 8; ++r) {
          int M = mt * 16 + r + hi * 8;
          sO[M * 256 + h * 32 + nt * 16 + lo] = (h16)o[mt][nt][r];
        }
  }

  __syncthreads();

  // ---------------- Output projection -> global ----------------
  {
    v8f f[8];
#pragma unroll
    for (int t = 0; t < 8; ++t) f[t] = v8f{};
    gemm256<true>(nullptr, sO + (rs * 16 + lo) * 256 + hi * 8,
                  wp, sW0, sW1, f, tid, lo, hi, ch);
#pragma unroll
    for (int nt = 0; nt < 8; ++nt) {
      int col = (ch * 8 + nt) * 16 + lo;
      float pbv = pb[col];
#pragma unroll
      for (int r = 0; r < 8; ++r) {
        int M = rs * 16 + r + hi * 8;
        out[((size_t)b * 64 + M) * 256 + col] = f[nt][r] + pbv;
      }
    }
  }
}

extern "C" void kernel_launch(void* const* d_in, const int* in_sizes, int n_in,
                              void* d_out, int out_size, void* d_ws, size_t ws_size,
                              hipStream_t stream) {
  (void)in_sizes; (void)n_in; (void)out_size; (void)ws_size;
  const float* x    = (const float*)d_in[0];
  const float* c    = (const float*)d_in[1];
  const float* mask = (const float*)d_in[2];
  const float* qw   = (const float*)d_in[3];
  const float* qb   = (const float*)d_in[4];
  const float* kvw  = (const float*)d_in[5];
  const float* kvb  = (const float*)d_in[6];
  const float* lsc  = (const float*)d_in[7];
  const float* rpb  = (const float*)d_in[8];
  const float* pw   = (const float*)d_in[9];
  const float* pb   = (const float*)d_in[10];
  float* out = (float*)d_out;
  h16* w16 = (h16*)d_ws;   // 262144 halves = 512 KB of scratch

  // Stage 1: one-shot f32 -> f16 weight conversion (stream-ordered).
  convert_weights<<<dim3(1024), dim3(256), 0, stream>>>(qw, kvw, pw, w16);

  // Stage 2: fused window attention, one workgroup per window.
  swinv2_window_attn<<<dim3(4096), dim3(256), 0, stream>>>(
      x, c, mask, w16, qb, kvb, lsc, rpb, pb, out);
}
